// SwinTransformerBlock_73126113182015
// MI455X (gfx1250) — compile-verified, matrix operand order checked by reading, not run on hardware
//
#include <hip/hip_runtime.h>
#include <hip/hip_bf16.h>

typedef __attribute__((ext_vector_type(16))) _Float16 v16h;
typedef __attribute__((ext_vector_type(8)))  _Float16 v8h;
typedef __attribute__((ext_vector_type(8)))  float    v8f;

// Problem constants
#define BQ   32
#define HH   56
#define WW   56
#define CC   256
#define NHD  8
#define WS   7
#define SHV  3
#define HD   32
#define NTOK 49
#define NWIN 2048            // B * (56/7)*(56/7) = 32*64
#define MTOT 100352          // 2048*49 = 32*56*56
#define MLPC 1024
#define SCALE_F 0.17677669529663687f

// ---------------------------------------------------------------------------
// Weight transpose + f32 -> f16 convert:  dst[n*K + k] = src[k*N + n]
// ---------------------------------------------------------------------------
__global__ void wt_transpose_f16(_Float16* __restrict__ dst,
                                 const float* __restrict__ src,
                                 int K, int N) {
  int i = blockIdx.x * blockDim.x + threadIdx.x;
  if (i >= K * N) return;
  int k = i / N;
  int n = i - k * N;
  dst[(size_t)n * K + k] = (_Float16)src[i];
}

// ---------------------------------------------------------------------------
// LN1 + cyclic shift (-3,-3) + window partition -> f16 (window-major rows)
// one wave (32 lanes) per token, 8 tokens per block
// ---------------------------------------------------------------------------
__global__ __launch_bounds__(256) void ln1_shift_part(
    const float* __restrict__ x, const float* __restrict__ g,
    const float* __restrict__ be, _Float16* __restrict__ out) {
  int token = blockIdx.x * 8 + (threadIdx.x >> 5);
  int lane  = threadIdx.x & 31;
  int win = token / NTOK;
  int t   = token - win * NTOK;
  int b   = win >> 6;
  int wv  = win & 63;
  int wi  = wv >> 3, wj = wv & 7;
  int i   = t / WS, j = t - i * WS;
  int hs  = wi * WS + i, ws = wj * WS + j;
  int h = hs + SHV; if (h >= HH) h -= HH;
  int w = ws + SHV; if (w >= WW) w -= WW;
  const float* px = x + (((size_t)b * HH + h) * WW + w) * CC + lane * 8;
  float4 p0 = *(const float4*)(px);
  float4 p1 = *(const float4*)(px + 4);
  float v[8] = {p0.x, p0.y, p0.z, p0.w, p1.x, p1.y, p1.z, p1.w};
  float s = 0.f, s2 = 0.f;
  for (int e = 0; e < 8; ++e) { s += v[e]; s2 += v[e] * v[e]; }
  for (int off = 1; off < 32; off <<= 1) {
    s  += __shfl_xor(s, off, 32);
    s2 += __shfl_xor(s2, off, 32);
  }
  float mean = s * (1.0f / CC);
  float var  = s2 * (1.0f / CC) - mean * mean;
  float inv  = rsqrtf(var + 1e-5f);
  const float* gp = g + lane * 8;
  const float* bp = be + lane * 8;
  v8h o;
  for (int e = 0; e < 8; ++e)
    o[e] = (_Float16)((v[e] - mean) * inv * gp[e] + bp[e]);
  *(v8h*)(out + (size_t)token * CC + lane * 8) = o;
}

// ---------------------------------------------------------------------------
// LN2: natural token order, reads f32 residual (d_out), writes f16
// ---------------------------------------------------------------------------
__global__ __launch_bounds__(256) void ln2_kernel(
    const float* __restrict__ x, const float* __restrict__ g,
    const float* __restrict__ be, _Float16* __restrict__ out) {
  int token = blockIdx.x * 8 + (threadIdx.x >> 5);
  int lane  = threadIdx.x & 31;
  const float* px = x + (size_t)token * CC + lane * 8;
  float4 p0 = *(const float4*)(px);
  float4 p1 = *(const float4*)(px + 4);
  float v[8] = {p0.x, p0.y, p0.z, p0.w, p1.x, p1.y, p1.z, p1.w};
  float s = 0.f, s2 = 0.f;
  for (int e = 0; e < 8; ++e) { s += v[e]; s2 += v[e] * v[e]; }
  for (int off = 1; off < 32; off <<= 1) {
    s  += __shfl_xor(s, off, 32);
    s2 += __shfl_xor(s2, off, 32);
  }
  float mean = s * (1.0f / CC);
  float var  = s2 * (1.0f / CC) - mean * mean;
  float inv  = rsqrtf(var + 1e-5f);
  const float* gp = g + lane * 8;
  const float* bp = be + lane * 8;
  v8h o;
  for (int e = 0; e < 8; ++e)
    o[e] = (_Float16)((v[e] - mean) * inv * gp[e] + bp[e]);
  *(v8h*)(out + (size_t)token * CC + lane * 8) = o;
}

// ---------------------------------------------------------------------------
// Generic WMMA GEMM: C(M,N) = A(M,K) * Bt(N,K)^T + bias, fused epilogues.
// Block tile 128x128, 8 waves (4x2), wave tile 32x64 (2x4 WMMA tiles), BK=32.
// Double-buffered LDS, register-staged global loads, one barrier per K-step.
// EPI: 0 = relu -> f16        (QKV)
//      1 = reverse-shift + residual -> f32 d_out (proj)
//      2 = exact GELU -> f16  (FC1)
//      3 = accumulate into f32 d_out (FC2 + residual)
// ---------------------------------------------------------------------------
#define GBM 128
#define GBN 128
#define GBK 32

template <int EPI>
__global__ __launch_bounds__(256) void gemm_wmma(
    const _Float16* __restrict__ A, const _Float16* __restrict__ Bt,
    const float* __restrict__ bias, const float* __restrict__ aux,
    void* __restrict__ outp, int M, int N, int K) {
  __shared__ _Float16 sA[2][GBM][GBK];
  __shared__ _Float16 sB[2][GBN][GBK];
  const int bm  = blockIdx.x * GBM;
  const int bn  = blockIdx.y * GBN;
  const int tid = threadIdx.x;
  const int wvi = tid >> 5;
  const int lane = tid & 31;
  const int wm = wvi >> 1;          // 0..3  (M strip of 32)
  const int wn = wvi & 1;           // 0..1  (N strip of 64)
  const int lhi = lane >> 4;        // K-half selector
  const int llo = lane & 15;        // row/col within tile

  v8f acc[2][4];
  for (int a = 0; a < 2; ++a)
    for (int b = 0; b < 4; ++b) acc[a][b] = (v8f){};

  const int lrow = tid >> 1;            // 0..127
  const int lseg = (tid & 1) * 16;      // 0 / 16

  const _Float16* pa = &A[(size_t)(bm + lrow) * K + lseg];
  const _Float16* pb = &Bt[(size_t)(bn + lrow) * K + lseg];

  // preload first K tile
  v16h ga = *(const v16h*)pa;
  v16h gb = *(const v16h*)pb;
  *(v16h*)&sA[0][lrow][lseg] = ga;
  *(v16h*)&sB[0][lrow][lseg] = gb;
  __syncthreads();

  int cur = 0;
  for (int k0 = 0; k0 < K; k0 += GBK) {
    const int nk = k0 + GBK;
    if (nk < K) {
      ga = *(const v16h*)(pa + nk);
      gb = *(const v16h*)(pb + nk);
    }
    v16h af[2], bf[4];
    for (int tm = 0; tm < 2; ++tm)
      af[tm] = *(const v16h*)&sA[cur][wm * 32 + tm * 16 + llo][lhi * 16];
    for (int tn = 0; tn < 4; ++tn)
      bf[tn] = *(const v16h*)&sB[cur][wn * 64 + tn * 16 + llo][lhi * 16];
    for (int tm = 0; tm < 2; ++tm)
      for (int tn = 0; tn < 4; ++tn)
        acc[tm][tn] = __builtin_amdgcn_wmma_f32_16x16x32_f16(
            false, af[tm], false, bf[tn], (short)0, acc[tm][tn], false, false);
    if (nk < K) {
      const int nxt = cur ^ 1;
      *(v16h*)&sA[nxt][lrow][lseg] = ga;
      *(v16h*)&sB[nxt][lrow][lseg] = gb;
      __syncthreads();
      cur = nxt;
    }
  }

  for (int tm = 0; tm < 2; ++tm)
    for (int tn = 0; tn < 4; ++tn)
      for (int r = 0; r < 8; ++r) {
        int row = bm + wm * 32 + tm * 16 + r + 8 * lhi;
        int col = bn + wn * 64 + tn * 16 + llo;
        float val = acc[tm][tn][r] + bias[col];
        if constexpr (EPI == 0) {
          ((_Float16*)outp)[(size_t)row * N + col] = (_Float16)fmaxf(val, 0.f);
        } else if constexpr (EPI == 1) {
          // rows are window-major tokens of the shifted image
          int win = row / NTOK;
          int t   = row - win * NTOK;
          int b   = win >> 6;
          int wv2 = win & 63;
          int wi  = wv2 >> 3, wj = wv2 & 7;
          int i   = t / WS, j = t - i * WS;
          int h = wi * WS + i + SHV; if (h >= HH) h -= HH;
          int w = wj * WS + j + SHV; if (w >= WW) w -= WW;
          size_t idx = (((size_t)b * HH + h) * WW + w) * CC + col;
          ((float*)outp)[idx] = aux[idx] + val;
        } else if constexpr (EPI == 2) {
          float gl = 0.5f * val * (1.0f + erff(val * 0.70710678118654752f));
          ((_Float16*)outp)[(size_t)row * N + col] = (_Float16)gl;
        } else {
          size_t idx = (size_t)row * N + col;
          ((float*)outp)[idx] = ((float*)outp)[idx] + val;
        }
      }
}

// ---------------------------------------------------------------------------
// Windowed attention: one block per (window, head). 4 waves, 128 threads.
// S = relu(scale*QK^T + rel_pos_bias + shift_mask); softmax; O = relu(S*V)
// ---------------------------------------------------------------------------
__device__ __forceinline__ int region_grp(int c) {
  return (c < HH - WS) ? 0 : ((c < HH - SHV) ? 1 : 2);
}

__global__ __launch_bounds__(128) void attn_kernel(
    const _Float16* __restrict__ qkv, const float* __restrict__ rpb,
    _Float16* __restrict__ att) {
  __shared__ _Float16 sQ[64][32];
  __shared__ _Float16 sK[64][32];
  __shared__ _Float16 sVT[32][64];
  __shared__ _Float16 sP[64][64];

  const int win  = blockIdx.x;
  const int head = blockIdx.y;
  const int tid  = threadIdx.x;

  // ---- stage Q, K, V^T into LDS (tokens padded 49 -> 64 with zeros) ----
  {
    int t = tid >> 1, half = tid & 1;
    const _Float16* base = qkv + ((size_t)win * NTOK) * 768 + head * HD;
    if (t < NTOK) {
      v16h qv = *(const v16h*)(base + (size_t)t * 768 + half * 16);
      v16h kv = *(const v16h*)(base + (size_t)t * 768 + 256 + half * 16);
      v16h vv = *(const v16h*)(base + (size_t)t * 768 + 512 + half * 16);
      *(v16h*)&sQ[t][half * 16] = qv;
      *(v16h*)&sK[t][half * 16] = kv;
      for (int e = 0; e < 16; ++e) sVT[half * 16 + e][t] = vv[e];
    } else {
      v16h z = (v16h){};
      *(v16h*)&sQ[t][half * 16] = z;
      *(v16h*)&sK[t][half * 16] = z;
      for (int e = 0; e < 16; ++e) sVT[half * 16 + e][t] = (_Float16)0.f;
    }
  }
  __syncthreads();

  const int wv  = tid >> 5;    // wave id: row strip [16*wv, 16*wv+16)
  const int lane = tid & 31;
  const int lhi = lane >> 4, llo = lane & 15;

  // ---- S = Q * K^T  (64x64 via 4x4 tiles, this wave does one row strip) ----
  v16h aq = *(const v16h*)&sQ[wv * 16 + llo][lhi * 16];
  v8f s[4];
  for (int tn = 0; tn < 4; ++tn) {
    v16h bk = *(const v16h*)&sK[tn * 16 + llo][lhi * 16];
    v8f z = (v8f){};
    s[tn] = __builtin_amdgcn_wmma_f32_16x16x32_f16(
        false, aq, false, bk, (short)0, z, false, false);
  }

  // ---- scale + rel-pos bias + shifted-window mask + relu + softmax ----
  const int wi = (win & 63) >> 3, wj = win & 7;
  for (int r = 0; r < 8; ++r) {
    int a = wv * 16 + r + 8 * lhi;
    int ia = a / WS, ja = a - ia * WS;
    float vals[4];
    for (int tn = 0; tn < 4; ++tn) {
      int bcol = tn * 16 + llo;
      float sv = s[tn][r] * SCALE_F;
      if (a < NTOK && bcol < NTOK) {
        int ib = bcol / WS, jb = bcol - ib * WS;
        int ridx = (ia - ib + WS - 1) * (2 * WS - 1) + (ja - jb + WS - 1);
        float bias = rpb[ridx * NHD + head];
        int ga = region_grp(wi * WS + ia) * 3 + region_grp(wj * WS + ja);
        int gb = region_grp(wi * WS + ib) * 3 + region_grp(wj * WS + jb);
        sv = sv + bias + ((ga == gb) ? 0.f : -1e9f);
        sv = fmaxf(sv, 0.f);
      } else {
        sv = -1e30f;
      }
      vals[tn] = sv;
    }
    float m = fmaxf(fmaxf(vals[0], vals[1]), fmaxf(vals[2], vals[3]));
    for (int off = 1; off < 16; off <<= 1) m = fmaxf(m, __shfl_xor(m, off, 32));
    float e[4], esum = 0.f;
    for (int tn = 0; tn < 4; ++tn) {
      e[tn] = (vals[tn] > -1e29f) ? __expf(vals[tn] - m) : 0.f;
      esum += e[tn];
    }
    for (int off = 1; off < 16; off <<= 1) esum += __shfl_xor(esum, off, 32);
    float inv = (esum > 0.f) ? (1.0f / esum) : 0.f;
    for (int tn = 0; tn < 4; ++tn)
      sP[a][tn * 16 + llo] = (_Float16)(e[tn] * inv);
  }
  __syncthreads();

  // ---- O = P * V   (64x32, K=64 -> 2 K-steps, 2 col tiles) ----
  v8f o[2] = {(v8f){}, (v8f){}};
  for (int kk = 0; kk < 2; ++kk) {
    v16h ap = *(const v16h*)&sP[wv * 16 + llo][kk * 32 + lhi * 16];
    for (int tn = 0; tn < 2; ++tn) {
      v16h bv = *(const v16h*)&sVT[tn * 16 + llo][kk * 32 + lhi * 16];
      o[tn] = __builtin_amdgcn_wmma_f32_16x16x32_f16(
          false, ap, false, bv, (short)0, o[tn], false, false);
    }
  }
  for (int tn = 0; tn < 2; ++tn)
    for (int r = 0; r < 8; ++r) {
      int a = wv * 16 + r + 8 * lhi;
      if (a < NTOK)
        att[((size_t)win * NTOK + a) * CC + head * HD + tn * 16 + llo] =
            (_Float16)fmaxf(o[tn][r], 0.f);
    }
}

// ---------------------------------------------------------------------------
// Launch
// ---------------------------------------------------------------------------
extern "C" void kernel_launch(void* const* d_in, const int* in_sizes, int n_in,
                              void* d_out, int out_size, void* d_ws, size_t ws_size,
                              hipStream_t stream) {
  const float* x      = (const float*)d_in[0];
  const float* gamma1 = (const float*)d_in[1];
  const float* beta1  = (const float*)d_in[2];
  const float* w_qkv  = (const float*)d_in[3];
  const float* b_qkv  = (const float*)d_in[4];
  const float* rpb    = (const float*)d_in[5];
  const float* w_proj = (const float*)d_in[6];
  const float* b_proj = (const float*)d_in[7];
  const float* gamma2 = (const float*)d_in[8];
  const float* beta2  = (const float*)d_in[9];
  const float* w_fc1  = (const float*)d_in[10];
  const float* b_fc1  = (const float*)d_in[11];
  const float* w_fc2  = (const float*)d_in[12];
  const float* b_fc2  = (const float*)d_in[13];
  float* outf = (float*)d_out;

  // workspace layout (bytes, 256-aligned)
  char* ws = (char*)d_ws;
  _Float16* wqkvT = (_Float16*)(ws + 0);                       //  768*256 f16
  _Float16* wprojT = (_Float16*)(ws + 393216);                 //  256*256 f16
  _Float16* wfc1T = (_Float16*)(ws + 524288);                  // 1024*256 f16
  _Float16* wfc2T = (_Float16*)(ws + 1048576);                 //  256*1024 f16
  _Float16* buf1  = (_Float16*)(ws + 1572864);                 // M*256 f16 (xw16 / h16)
  _Float16* buf2  = (_Float16*)(ws + 1572864 + (size_t)MTOT * CC * 2);
  _Float16* qkv16 = buf2;                                      // M*768 f16
  _Float16* att16 = buf2 + (size_t)MTOT * 768;                 // M*256 f16
  _Float16* g16   = buf2;                                      // M*1024 f16 (reuses)

  // 1) weight prep
  wt_transpose_f16<<<(256 * 768 + 255) / 256, 256, 0, stream>>>(wqkvT, w_qkv, 256, 768);
  wt_transpose_f16<<<(256 * 256 + 255) / 256, 256, 0, stream>>>(wprojT, w_proj, 256, 256);
  wt_transpose_f16<<<(256 * 1024 + 255) / 256, 256, 0, stream>>>(wfc1T, w_fc1, 256, 1024);
  wt_transpose_f16<<<(1024 * 256 + 255) / 256, 256, 0, stream>>>(wfc2T, w_fc2, 1024, 256);

  // 2) LN1 + shift + partition
  ln1_shift_part<<<MTOT / 8, 256, 0, stream>>>(x, gamma1, beta1, buf1);

  // 3) QKV projection + relu gates
  gemm_wmma<0><<<dim3(MTOT / GBM, 768 / GBN), 256, 0, stream>>>(
      buf1, wqkvT, b_qkv, nullptr, qkv16, MTOT, 768, 256);

  // 4) windowed attention
  attn_kernel<<<dim3(NWIN, NHD), 128, 0, stream>>>(qkv16, rpb, att16);

  // 5) output projection + reverse shift + residual  (d_out = x_res)
  gemm_wmma<1><<<dim3(MTOT / GBM, 256 / GBN), 256, 0, stream>>>(
      att16, wprojT, b_proj, x, d_out, MTOT, 256, 256);

  // 6) LN2
  ln2_kernel<<<MTOT / 8, 256, 0, stream>>>(outf, gamma2, beta2, buf1);

  // 7) FC1 + exact GELU
  gemm_wmma<2><<<dim3(MTOT / GBM, 1024 / GBN), 256, 0, stream>>>(
      buf1, wfc1T, b_fc1, nullptr, g16, MTOT, 1024, 256);

  // 8) FC2 + residual accumulate into d_out
  gemm_wmma<3><<<dim3(MTOT / GBM, 256 / GBN), 256, 0, stream>>>(
      g16, wfc2T, b_fc2, nullptr, d_out, MTOT, 256, 1024);
}
